// MultiHeadAttention_37271726195009
// MI455X (gfx1250) — matrix operand order repaired and checked by reading, not compiled, not run on hardware
//
#include <hip/hip_runtime.h>
#include <hip/hip_bf16.h>
#include <math.h>

// ---------------------------------------------------------------------------
// MultiHeadAttention fused bf16-WMMA pipeline for gfx1250 (MI455X)
//   B=4, S=2048, HIDDEN=1024, HEADS=16, HEAD_DIM=64
// Pipeline:
//   0) cvt: X, Wq, Wk, Wv, Wp  fp32 -> bf16 (one-time, removes cvt from hot loops)
//   1) qkv: X@W^T+b via v_wmma_f32_16x16x32_bf16, software-pipelined; V stored ^T
//   2) attn: flash attention, online softmax, P staged through per-wave LDS
//   3) proj: ctx@Wp^T+bp, * hidden_z
// ---------------------------------------------------------------------------

#define HIDDEN   1024
#define HEADS    16
#define HEAD_DIM 64
#define BATCH    4
#define SEQ      2048
#define ROWS     (BATCH * SEQ)        // 8192

typedef __bf16 bf16;
typedef __attribute__((ext_vector_type(16))) __bf16 v16bf;
typedef __attribute__((ext_vector_type(8)))  float  v8f;

union ABf {            // one WMMA A/B fragment: 16 bf16 = 8 VGPRs = 2x uint4
    v16bf v;
    uint4 u2[2];
};

__device__ __forceinline__ v8f wmma_bf16(v16bf a, v16bf b, v8f c) {
    return __builtin_amdgcn_wmma_f32_16x16x32_bf16(
        /*neg_a=*/false, a, /*neg_b=*/false, b,
        /*c_mod=*/(short)0, c, /*reuse_a=*/false, /*reuse_b=*/false);
}

// A fragment (16x32, M x K) from bf16 row-major [., ld].
// Lane layout: lane&15 = M; lane>>4 selects K octet {0-7 | 8-15}; elems 8..15 = K+16.
__device__ __forceinline__ v16bf load_a_bf16(const bf16* X, int ld, int row0, int k0, int lane) {
    const int m = lane & 15, koff = (lane >> 4) << 3;
    const bf16* p = X + (size_t)(row0 + m) * ld + k0 + koff;
    ABf a;
    a.u2[0] = *(const uint4*)(p);
    a.u2[1] = *(const uint4*)(p + 16);
    return a.v;
}

// B fragment (32x16, K x N) with B[k][n] = SRC[n0+n][k0+k]: lane&15 = N,
// lane>>4 selects K octet {0-7 | 8-15}; elems 8..15 = K+16 (2x16B).
__device__ __forceinline__ v16bf load_b_bf16(const bf16* S, int ld, int n0, int k0, int lane) {
    const int n = lane & 15, kk = (lane >> 4) << 3;
    const bf16* p = S + (size_t)(n0 + n) * ld + k0 + kk;
    ABf b;
    b.u2[0] = *(const uint4*)(p);
    b.u2[1] = *(const uint4*)(p + 16);
    return b.v;
}

// ---------------------------------------------------------------------------
// Kernel 0: fp32 -> bf16 conversion (one thread per group of 4)
// ---------------------------------------------------------------------------
__global__ void __launch_bounds__(256)
mha_cvt_kernel(const float* __restrict__ src, bf16* __restrict__ dst, int n4) {
    int i = blockIdx.x * blockDim.x + threadIdx.x;
    if (i < n4) {
        float4 f = ((const float4*)src)[i];
        bf16* o = dst + 4 * (size_t)i;
        o[0] = (bf16)f.x; o[1] = (bf16)f.y; o[2] = (bf16)f.z; o[3] = (bf16)f.w;
    }
}

// ---------------------------------------------------------------------------
// Kernel 1: QKV projection.  grid = (1024, 1, 3), block = 256 (8 waves).
// Each wave: one 16-row x 64-col tile of X @ W^T + b, software-pipelined.
//   z=0 -> Q (scaled by 1/sqrt(64)), z=1 -> K, z=2 -> V transposed [B,HIDDEN,S]
// ---------------------------------------------------------------------------
__global__ void __launch_bounds__(256)
mha_qkv_kernel(const bf16* __restrict__ Xb,
               const bf16* __restrict__ Wqb, const float* __restrict__ bq,
               const bf16* __restrict__ Wkb, const float* __restrict__ bk,
               const bf16* __restrict__ Wvb, const float* __restrict__ bv,
               bf16* __restrict__ Qb, bf16* __restrict__ Kb, bf16* __restrict__ Vt) {
    const int lane = threadIdx.x & 31;
    const int wave = threadIdx.x >> 5;
    const int tile = blockIdx.x * 8 + wave;         // 8192 wave-tiles
    const int row0 = (tile >> 4) * 16;              // 512 row strips
    const int col0 = (tile & 15) * 64;              // 16 col groups of 64

    const int z = blockIdx.z;
    const bf16*  W    = (z == 0) ? Wqb : (z == 1) ? Wkb : Wvb;
    const float* bias = (z == 0) ? bq  : (z == 1) ? bk  : bv;

    v8f acc0 = {}, acc1 = {}, acc2 = {}, acc3 = {};
    // software pipeline: fragments for step kb live while step kb+32 loads issue
    v16bf a  = load_a_bf16(Xb, HIDDEN, row0, 0, lane);
    v16bf b0 = load_b_bf16(W, HIDDEN, col0 + 0,  0, lane);
    v16bf b1 = load_b_bf16(W, HIDDEN, col0 + 16, 0, lane);
    v16bf b2 = load_b_bf16(W, HIDDEN, col0 + 32, 0, lane);
    v16bf b3 = load_b_bf16(W, HIDDEN, col0 + 48, 0, lane);
    for (int kb = 0; kb < HIDDEN; kb += 32) {
        const int kn = (kb + 32 < HIDDEN) ? kb + 32 : 0;   // wrap: harmless reload
        // prefetch 1 step ahead (global_prefetch_b8, speculative)
        __builtin_prefetch(Xb + (size_t)(row0 + (lane & 15)) * HIDDEN + kn, 0, 0);
        v16bf an  = load_a_bf16(Xb, HIDDEN, row0, kn, lane);
        v16bf bn0 = load_b_bf16(W, HIDDEN, col0 + 0,  kn, lane);
        v16bf bn1 = load_b_bf16(W, HIDDEN, col0 + 16, kn, lane);
        v16bf bn2 = load_b_bf16(W, HIDDEN, col0 + 32, kn, lane);
        v16bf bn3 = load_b_bf16(W, HIDDEN, col0 + 48, kn, lane);
        acc0 = wmma_bf16(a, b0, acc0);
        acc1 = wmma_bf16(a, b1, acc1);
        acc2 = wmma_bf16(a, b2, acc2);
        acc3 = wmma_bf16(a, b3, acc3);
        a = an; b0 = bn0; b1 = bn1; b2 = bn2; b3 = bn3;
    }

    const int half = lane >> 4;
    v8f* accs[4] = { &acc0, &acc1, &acc2, &acc3 };
#pragma unroll
    for (int t = 0; t < 4; ++t) {
        const int col = col0 + t * 16 + (lane & 15);
        const float bv_ = bias[col];
#pragma unroll
        for (int r = 0; r < 8; ++r) {
            const int row = row0 + r + 8 * half;
            float val = (*accs[t])[r] + bv_;
            if (z == 0) {
                Qb[(size_t)row * HIDDEN + col] = (bf16)(val * 0.125f); // 1/sqrt(64)
            } else if (z == 1) {
                Kb[(size_t)row * HIDDEN + col] = (bf16)val;
            } else {
                const int b = row >> 11, s = row & (SEQ - 1);
                Vt[((size_t)(b * HIDDEN + col)) * SEQ + s] = (bf16)val;
            }
        }
    }
}

// ---------------------------------------------------------------------------
// Kernel 2: fused flash attention.  grid = (16, 64), block = 256 (8 waves).
// blockIdx.y = b*HEADS + h; each wave owns 16 query rows, streams 32 keys/iter.
// V fragments issue at loop top; next K fragments issue before the softmax so
// VMEM overlaps the exp/shuffle VALU work and the LDS transpose round-trip.
// ---------------------------------------------------------------------------
__global__ void __launch_bounds__(256)
mha_attn_kernel(const bf16* __restrict__ Qb, const bf16* __restrict__ Kb,
                const bf16* __restrict__ Vt, const float* __restrict__ heads_z,
                bf16* __restrict__ Ctx) {
    const int lane = threadIdx.x & 31;
    const int wave = threadIdx.x >> 5;
    const int bh = blockIdx.y;
    const int b = bh >> 4, h = bh & 15;
    const int hoff = h * HEAD_DIM;
    const int qrow = b * SEQ + blockIdx.x * 128 + wave * 16;   // global Q row base
    const int half = lane >> 4;
    const int vrow = b * HIDDEN + hoff;

    // Per-wave LDS staging for P (16 rows x 32 keys), row stride 40 bf16 (bank-spread).
    __shared__ __align__(16) bf16 Pbuf[8][16 * 40];
    bf16* Pw = &Pbuf[wave][0];

    // Q fragments (reused across all key blocks): K-dim = 64 -> two 16x32 frags.
    const v16bf qa0 = load_a_bf16(Qb, HIDDEN, qrow, hoff + 0,  lane);
    const v16bf qa1 = load_a_bf16(Qb, HIDDEN, qrow, hoff + 32, lane);

    v8f acc0 = {}, acc1 = {}, acc2 = {}, acc3 = {};
    float m_run[8], l_run[8];
#pragma unroll
    for (int r = 0; r < 8; ++r) { m_run[r] = -3.0e38f; l_run[r] = 0.0f; }

    // preload first key block's K fragments
    v16bf kf0 = load_b_bf16(Kb, HIDDEN, b * SEQ + 0,  hoff + 0,  lane);
    v16bf kf1 = load_b_bf16(Kb, HIDDEN, b * SEQ + 0,  hoff + 32, lane);
    v16bf kf2 = load_b_bf16(Kb, HIDDEN, b * SEQ + 16, hoff + 0,  lane);
    v16bf kf3 = load_b_bf16(Kb, HIDDEN, b * SEQ + 16, hoff + 32, lane);

    for (int jb = 0; jb < SEQ; jb += 32) {
        // V fragments for this block: independent of scores, issue first
        v16bf vf0 = load_b_bf16(Vt, SEQ, vrow + 0,  jb, lane);
        v16bf vf1 = load_b_bf16(Vt, SEQ, vrow + 16, jb, lane);
        v16bf vf2 = load_b_bf16(Vt, SEQ, vrow + 32, jb, lane);
        v16bf vf3 = load_b_bf16(Vt, SEQ, vrow + 48, jb, lane);

        // scores: two 16x16 tiles (keys jb..+15, jb+16..+31), fp32 accum
        v8f s0 = {}, s1 = {};
        s0 = wmma_bf16(qa0, kf0, s0);
        s0 = wmma_bf16(qa1, kf1, s0);
        s1 = wmma_bf16(qa0, kf2, s1);
        s1 = wmma_bf16(qa1, kf3, s1);

        // preload next key block's K fragments (overlaps softmax + LDS)
        const int jn = (jb + 32 < SEQ) ? jb + 32 : 0;      // wrap: harmless reload
        const int krown = b * SEQ + jn;
        kf0 = load_b_bf16(Kb, HIDDEN, krown,      hoff + 0,  lane);
        kf1 = load_b_bf16(Kb, HIDDEN, krown,      hoff + 32, lane);
        kf2 = load_b_bf16(Kb, HIDDEN, krown + 16, hoff + 0,  lane);
        kf3 = load_b_bf16(Kb, HIDDEN, krown + 16, hoff + 32, lane);

        // online softmax per row; masks 1/2/4/8 reduce within each 16-lane half,
        // matching C-layout rows (lanes 0-15: M=r, lanes 16-31: M=8+r)
        const int col = lane & 15;
#pragma unroll
        for (int r = 0; r < 8; ++r) {
            float v = fmaxf(s0[r], s1[r]);
            v = fmaxf(v, __shfl_xor(v, 1, 32));
            v = fmaxf(v, __shfl_xor(v, 2, 32));
            v = fmaxf(v, __shfl_xor(v, 4, 32));
            v = fmaxf(v, __shfl_xor(v, 8, 32));
            const float mnew  = fmaxf(m_run[r], v);
            const float alpha = __expf(m_run[r] - mnew);
            const float p0 = __expf(s0[r] - mnew);
            const float p1 = __expf(s1[r] - mnew);
            float rs = p0 + p1;
            rs += __shfl_xor(rs, 1, 32);
            rs += __shfl_xor(rs, 2, 32);
            rs += __shfl_xor(rs, 4, 32);
            rs += __shfl_xor(rs, 8, 32);
            l_run[r] = l_run[r] * alpha + rs;
            m_run[r] = mnew;
            acc0[r] *= alpha; acc1[r] *= alpha; acc2[r] *= alpha; acc3[r] *= alpha;
            const int row = r + 8 * half;
            Pw[row * 40 + col]      = (bf16)p0;
            Pw[row * 40 + 16 + col] = (bf16)p1;
        }

        // C-layout -> A-layout transpose through wave-local LDS (DS ops in-order)
        ABf pa;
        {
            const int m = lane & 15, koff = (lane >> 4) << 3;
            pa.u2[0] = *(const uint4*)(Pw + m * 40 + koff);
            pa.u2[1] = *(const uint4*)(Pw + m * 40 + 16 + koff);
        }

        // ctx += P(16x32 keys) x V(32 keys x 64 dims)
        acc0 = wmma_bf16(pa.v, vf0, acc0);
        acc1 = wmma_bf16(pa.v, vf1, acc1);
        acc2 = wmma_bf16(pa.v, vf2, acc2);
        acc3 = wmma_bf16(pa.v, vf3, acc3);
    }

    const float hz = heads_z[h];
    v8f* accs[4] = { &acc0, &acc1, &acc2, &acc3 };
#pragma unroll
    for (int r = 0; r < 8; ++r) {
        const float inv = hz / l_run[r];
        const int row = qrow + r + 8 * half;
#pragma unroll
        for (int t = 0; t < 4; ++t) {
            const int col = hoff + t * 16 + (lane & 15);
            Ctx[(size_t)row * HIDDEN + col] = (bf16)((*accs[t])[r] * inv);
        }
    }
}

// ---------------------------------------------------------------------------
// Kernel 3: output projection.  grid = 1024, block = 256 (8 waves).
// out = (ctx @ Wp^T + bp) * hidden_z, fp32 output; same pipelined structure.
// ---------------------------------------------------------------------------
__global__ void __launch_bounds__(256)
mha_proj_kernel(const bf16* __restrict__ Ctx, const bf16* __restrict__ Wpb,
                const float* __restrict__ bp, const float* __restrict__ hz,
                float* __restrict__ Out) {
    const int lane = threadIdx.x & 31;
    const int wave = threadIdx.x >> 5;
    const int tile = blockIdx.x * 8 + wave;
    const int row0 = (tile >> 4) * 16;
    const int col0 = (tile & 15) * 64;

    v8f acc0 = {}, acc1 = {}, acc2 = {}, acc3 = {};
    v16bf a  = load_a_bf16(Ctx, HIDDEN, row0, 0, lane);
    v16bf b0 = load_b_bf16(Wpb, HIDDEN, col0 + 0,  0, lane);
    v16bf b1 = load_b_bf16(Wpb, HIDDEN, col0 + 16, 0, lane);
    v16bf b2 = load_b_bf16(Wpb, HIDDEN, col0 + 32, 0, lane);
    v16bf b3 = load_b_bf16(Wpb, HIDDEN, col0 + 48, 0, lane);
    for (int kb = 0; kb < HIDDEN; kb += 32) {
        const int kn = (kb + 32 < HIDDEN) ? kb + 32 : 0;
        __builtin_prefetch(Ctx + (size_t)(row0 + (lane & 15)) * HIDDEN + kn, 0, 0);
        v16bf an  = load_a_bf16(Ctx, HIDDEN, row0, kn, lane);
        v16bf bn0 = load_b_bf16(Wpb, HIDDEN, col0 + 0,  kn, lane);
        v16bf bn1 = load_b_bf16(Wpb, HIDDEN, col0 + 16, kn, lane);
        v16bf bn2 = load_b_bf16(Wpb, HIDDEN, col0 + 32, kn, lane);
        v16bf bn3 = load_b_bf16(Wpb, HIDDEN, col0 + 48, kn, lane);
        acc0 = wmma_bf16(a, b0, acc0);
        acc1 = wmma_bf16(a, b1, acc1);
        acc2 = wmma_bf16(a, b2, acc2);
        acc3 = wmma_bf16(a, b3, acc3);
        a = an; b0 = bn0; b1 = bn1; b2 = bn2; b3 = bn3;
    }

    const int half = lane >> 4;
    v8f* accs[4] = { &acc0, &acc1, &acc2, &acc3 };
#pragma unroll
    for (int t = 0; t < 4; ++t) {
        const int col = col0 + t * 16 + (lane & 15);
        const float bb = bp[col], zz = hz[col];
#pragma unroll
        for (int r = 0; r < 8; ++r) {
            const int row = row0 + r + 8 * half;
            Out[(size_t)row * HIDDEN + col] = ((*accs[t])[r] + bb) * zz;
        }
    }
}

// ---------------------------------------------------------------------------
extern "C" void kernel_launch(void* const* d_in, const int* in_sizes, int n_in,
                              void* d_out, int out_size, void* d_ws, size_t ws_size,
                              hipStream_t stream) {
    const float* X   = (const float*)d_in[0];
    const float* Wq  = (const float*)d_in[1];
    const float* bq  = (const float*)d_in[2];
    const float* Wk  = (const float*)d_in[3];
    const float* bk  = (const float*)d_in[4];
    const float* Wv  = (const float*)d_in[5];
    const float* bv  = (const float*)d_in[6];
    const float* Wp  = (const float*)d_in[7];
    const float* bp  = (const float*)d_in[8];
    const float* hdz = (const float*)d_in[9];   // heads_z  [16]
    const float* hiz = (const float*)d_in[10];  // hidden_z [1024]
    float* Out = (float*)d_out;

    const size_t MAT  = (size_t)ROWS * HIDDEN;    // 8.39M elems
    const size_t WMAT = (size_t)HIDDEN * HIDDEN;  // 1.05M elems
    bf16* Qb  = (bf16*)d_ws;
    bf16* Kb  = Qb  + MAT;
    bf16* Vt  = Kb  + MAT;
    bf16* Ctx = Vt  + MAT;
    bf16* Xb  = Ctx + MAT;
    bf16* Wqb = Xb  + MAT;
    bf16* Wkb = Wqb + WMAT;
    bf16* Wvb = Wkb + WMAT;
    bf16* Wpb = Wvb + WMAT;

    // 0) one-time fp32 -> bf16 conversions
    mha_cvt_kernel<<<dim3((int)(MAT / 4 / 256)), 256, 0, stream>>>(X,  Xb,  (int)(MAT / 4));
    mha_cvt_kernel<<<dim3((int)(WMAT / 4 / 256)), 256, 0, stream>>>(Wq, Wqb, (int)(WMAT / 4));
    mha_cvt_kernel<<<dim3((int)(WMAT / 4 / 256)), 256, 0, stream>>>(Wk, Wkb, (int)(WMAT / 4));
    mha_cvt_kernel<<<dim3((int)(WMAT / 4 / 256)), 256, 0, stream>>>(Wv, Wvb, (int)(WMAT / 4));
    mha_cvt_kernel<<<dim3((int)(WMAT / 4 / 256)), 256, 0, stream>>>(Wp, Wpb, (int)(WMAT / 4));

    mha_qkv_kernel<<<dim3(1024, 1, 3), 256, 0, stream>>>(
        Xb, Wqb, bq, Wkb, bk, Wvb, bv, Qb, Kb, Vt);
    mha_attn_kernel<<<dim3(16, 64), 256, 0, stream>>>(
        Qb, Kb, Vt, hdz, Ctx);
    mha_proj_kernel<<<dim3(1024), 256, 0, stream>>>(
        Ctx, Wpb, bp, hiz, Out);
}
